// RecurrentCell_64226940944748
// MI455X (gfx1250) — compile-verified, matrix operand order checked by reading, not run on hardware
//
#include <hip/hip_runtime.h>
#include <hip/hip_bf16.h>
#include <math.h>

// ---------------------------------------------------------------------------
// RecurrentCell (Elman RNN, tanh), B=64 T=512 I=H=512, fp32 in/out.
//   Phase 1: mem[b,t,:] = x[b,t,:] @ Wx + bx + bh        (f32 WMMA 16x16x4)
//   Phase 2: mem[b,t,:] = h_t = tanh(mem[b,t,:] + h_{t-1} @ Wh)
//            (f16 WMMA 16x16x32, f32 accumulate, persistent 4-block scan,
//             Wh fragments held in VGPRs for all 512 steps)
//   d_out = mem (64*512*512 floats) ++ hT (64*512 floats)
//   d_ws  = WhT in fp16, transposed [N=512][K=512]  -> 524288 bytes needed
// ---------------------------------------------------------------------------

typedef __attribute__((ext_vector_type(2)))  float     v2f;
typedef __attribute__((ext_vector_type(8)))  float     v8f;
typedef __attribute__((ext_vector_type(8)))  _Float16  v8h;
typedef __attribute__((ext_vector_type(16))) _Float16  v16h;

#define IH   512
#define TT   512
#define BB   64
#define HSTR 520   // padded LDS row stride (halves): 1040B -> 4-bank rotation/row

// V_TANH_F32 transcendental if the toolchain exposes it; ocml otherwise.
static __device__ __forceinline__ float fast_tanh(float x) {
#if __has_builtin(__builtin_amdgcn_tanhf)
    return __builtin_amdgcn_tanhf(x);
#else
    return tanhf(x);
#endif
}

// ---------------------------------------------------------------------------
// Kernel A: Wh (f32, [K][N] row-major) -> WhT (f16, [N][K] K-contiguous).
// ---------------------------------------------------------------------------
__global__ __launch_bounds__(256) void wh_cvt_tr(const float* __restrict__ Wh,
                                                 _Float16* __restrict__ WhT) {
    int idx = blockIdx.x * 256 + threadIdx.x;     // 0 .. 262143
    int n = idx >> 9;
    int k = idx & (IH - 1);
    WhT[idx] = (_Float16)Wh[k * IH + n];
}

// ---------------------------------------------------------------------------
// Kernel B: f32 GEMM  out[M=32768, N=512] = x @ Wx + (bx + bh)
// 256 threads = 8 waves; each wave: 16 rows x 64 cols, K stepped by 4 through
// V_WMMA_F32_16X16X4_F32. Grid: (32768/128) x (512/64) = 256 x 8.
// Memory floor ~129MB @ 23.3TB/s ≈ 5.5us; x (64MB) is L2-resident on reuse.
// ---------------------------------------------------------------------------
__global__ __launch_bounds__(256) void xw_gemm(const float* __restrict__ x,
                                               const float* __restrict__ Wx,
                                               const float* __restrict__ bx,
                                               const float* __restrict__ bh,
                                               float* __restrict__ out) {
    const int lane = threadIdx.x & 31;
    const int wave = threadIdx.x >> 5;
    const int m0   = blockIdx.x * 128 + wave * 16;
    const int n0   = blockIdx.y * 64;
    const int nl   = lane & 15;
    const int kSel = (lane & 16) ? 2 : 0;     // K phase for f32 A/B fragments
    const int rowHi= (lane & 16) ? 8 : 0;     // C/D row split across lane halves

    v8f z = {0.f, 0.f, 0.f, 0.f, 0.f, 0.f, 0.f, 0.f};
    v8f acc[4] = {z, z, z, z};

    const float* aptr = x + (size_t)(m0 + nl) * IH + kSel;  // A row = lane&15

#pragma unroll 2
    for (int k0 = 0; k0 < IH; k0 += 4) {
        v2f a = *(const v2f*)(aptr + k0);                    // K pair, 8B aligned
#pragma unroll
        for (int j = 0; j < 4; ++j) {
            const int n = n0 + j * 16 + nl;
            v2f b;
            b.x = Wx[(size_t)(k0 + kSel)     * IH + n];
            b.y = Wx[(size_t)(k0 + kSel + 1) * IH + n];
            acc[j] = __builtin_amdgcn_wmma_f32_16x16x4_f32(
                         false, a, false, b, (short)0, acc[j], false, false);
        }
    }

#pragma unroll
    for (int j = 0; j < 4; ++j) {
        const int n = n0 + j * 16 + nl;
        const float bias = bx[n] + bh[n];
#pragma unroll
        for (int r = 0; r < 8; ++r) {
            out[(size_t)(m0 + r + rowHi) * IH + n] = acc[j][r] + bias;
        }
    }
}

// ---------------------------------------------------------------------------
// Kernel C: persistent recurrent scan. Grid = 4 blocks (16 batch rows each),
// block = 1024 threads = 32 waves. Wave w owns output columns [16w, 16w+16).
// h lives in LDS (f16, double-buffered, padded stride); one s_barrier per
// timestep closes the recurrence. The wave's 16 Wh B-fragments (512x16 f16 =
// 128 VGPRs) are loaded ONCE and stay register-resident for all 512 steps,
// so the per-step chain is: ds_load A-frags -> 16x V_WMMA_F32_16X16X32_F16
// -> v_tanh -> ds_store -> s_barrier. xw_t is prefetched before the GEMM so
// its L2 latency hides under the WMMAs.
// ---------------------------------------------------------------------------
__global__ __launch_bounds__(1024) void rnn_scan(const _Float16* __restrict__ WhT,
                                                 float* __restrict__ out) {
    __shared__ __attribute__((aligned(16))) _Float16 hbuf[2][16][HSTR];

    const int tid = threadIdx.x;
    // h_0 = 0
    for (int i = tid; i < 2 * 16 * HSTR; i += 1024)
        ((_Float16*)hbuf)[i] = (_Float16)0.f;

    const int lane  = tid & 31;
    const int wave  = tid >> 5;                 // 0..31 -> N tile
    const int nl    = lane & 15;
    const int n     = wave * 16 + nl;           // this lane's output column
    const int koff  = (lane & 16) ? 8 : 0;      // f16 fragment K phase
    const int rA    = lane & 15;                // A-matrix row (batch-local)
    const int rowHi = (lane & 16) ? 8 : 0;      // C/D row split
    const int b0    = blockIdx.x * 16;          // batch tile base
    const size_t rowStride = (size_t)TT * IH;   // mem stride between batch rows

    // --- load this wave's Wh column slice into registers, once -------------
    const _Float16* bcol = WhT + (size_t)n * IH;
    v16h bfrag[16];
#pragma unroll
    for (int kt = 0; kt < 16; ++kt) {
        const int k0 = kt * 32 + koff;
        *(v8h*)&bfrag[kt]       = *(const v8h*)(bcol + k0);       // K: k0..k0+7
        *((v8h*)&bfrag[kt] + 1) = *(const v8h*)(bcol + k0 + 16);  // K: +16..+23
    }

    float* hT = out + (size_t)BB * TT * IH;
    const v8f z = {0.f, 0.f, 0.f, 0.f, 0.f, 0.f, 0.f, 0.f};

    __syncthreads();   // h_0 zero-init visible to all waves

    int p = 0;
    for (int t = 0; t < TT; ++t) {
        // Prefetch xw_t for this lane's 8 output elements (independent of LDS).
        const size_t off0 = ((size_t)(b0 + rowHi) * TT + t) * IH + n;
        float xwv[8];
#pragma unroll
        for (int r = 0; r < 8; ++r)
            xwv[r] = out[off0 + (size_t)r * rowStride];

        // GEMM: acc = h_{t-1} @ Wh[:, wave's 16 cols]
        v8f acc = z;
        const _Float16* arow = &hbuf[p][rA][0];
#pragma unroll
        for (int kt = 0; kt < 16; ++kt) {
            const int k0 = kt * 32 + koff;
            v16h a;
            *(v8h*)&a       = *(const v8h*)(arow + k0);        // K: k0..k0+7
            *((v8h*)&a + 1) = *(const v8h*)(arow + k0 + 16);   // K: k0+16..+23
            acc = __builtin_amdgcn_wmma_f32_16x16x32_f16(
                      false, a, false, bfrag[kt], (short)0, acc, false, false);
        }

        const int np = p ^ 1;
#pragma unroll
        for (int r = 0; r < 8; ++r) {
            const float v = fast_tanh(acc[r] + xwv[r]);
            out[off0 + (size_t)r * rowStride] = v;             // mem[b][t][n]
            hbuf[np][r + rowHi][n] = (_Float16)v;              // feed step t+1
            if (t == TT - 1)                                   // uniform branch
                hT[(size_t)(b0 + r + rowHi) * IH + n] = v;
        }
        __syncthreads();                                        // close recurrence
        p = np;
    }
}

// ---------------------------------------------------------------------------
extern "C" void kernel_launch(void* const* d_in, const int* in_sizes, int n_in,
                              void* d_out, int out_size, void* d_ws, size_t ws_size,
                              hipStream_t stream) {
    const float* x  = (const float*)d_in[0];   // [64,512,512]
    const float* Wx = (const float*)d_in[1];   // [512,512]
    const float* Wh = (const float*)d_in[2];   // [512,512]
    const float* bx = (const float*)d_in[3];   // [1,512]
    const float* bh = (const float*)d_in[4];   // [1,512]
    float* out = (float*)d_out;                // mem ++ hT
    _Float16* WhT = (_Float16*)d_ws;           // needs 512 KB scratch

    // 1) Wh -> transposed f16 copy (register/L2-resident operand for the scan)
    wh_cvt_tr<<<IH * IH / 256, 256, 0, stream>>>(Wh, WhT);
    // 2) xW = x@Wx + bx + bh, written straight into d_out's mem region
    xw_gemm<<<dim3((BB * TT) / 128, IH / 64), 256, 0, stream>>>(x, Wx, bx, bh, out);
    // 3) sequential scan, 4 persistent workgroups (one per 16-batch tile)
    rnn_scan<<<BB / 16, 1024, 0, stream>>>(WhT, out);
}
